// FastRCNN_87840671137913
// MI455X (gfx1250) — matrix operand order repaired and checked by reading, not compiled, not run on hardware
//
#include <hip/hip_runtime.h>
#include <hip/hip_bf16.h>

typedef __attribute__((ext_vector_type(16))) _Float16 v16h;
typedef __attribute__((ext_vector_type(8)))  float    v8f;

// ---------------- workspace layout (bytes, all 1KB aligned) ----------------
// A1sw : FC1 A fragments, f16, [16 mt][64 s][32 lane][16 slot]  = 1,048,576 B
// W1sw : FC1 B fragments, f16, [13 nt][64 s][32 lane][16 slot]  =   851,968 B
// A2sw : FC2 A fragments, f16, [16 mt][ 7 s][32 lane][16 slot]  =   114,688 B
// W2sw : FC2 B fragments, f16, [13 nt][ 7 s][32 lane][16 slot]  =    93,184 B
// F    : FC2 output, f32, [256][208]                            =   212,992 B
#define A1_OFF 0u
#define W1_OFF 1048576u
#define A2_OFF 1900544u
#define W2_OFF 2015232u
#define F_OFF  2108416u

static __device__ __forceinline__ int imin_(int a, int b) { return a < b ? a : b; }
static __device__ __forceinline__ int imax_(int a, int b) { return a > b ? a : b; }

// A-fragment (16-bit, 16x32 MxK) slot mapping per ISA 7.12.2:
// lane = half*16 + m ; half=(kk>>3)&1 ; slot=(kk&7)+((kk&16)?8:0)
static __device__ __forceinline__ void storeA(_Float16* A, int mt, int m, int k,
                                              int ksteps, float v) {
    int s = k >> 5, kk = k & 31;
    int hf = (kk >> 3) & 1;
    int slot = (kk & 7) + ((kk & 16) ? 8 : 0);
    A[(((mt * ksteps) + s) * 32 + hf * 16 + m) * 16 + slot] = (_Float16)v;
}

// B-fragment (16-bit, 32x16 KxN): lane = ((kk>>3)&1)*16 + n ; slot = (kk&7)+((kk&16)?8:0)
static __device__ __forceinline__ void storeB(_Float16* B, int nt, int n, int k,
                                              int ksteps, float v) {
    int s = k >> 5, kk = k & 31;
    B[(((nt * ksteps) + s) * 32 + ((kk >> 3) & 1) * 16 + n) * 16 + ((kk & 7) + ((kk & 16) ? 8 : 0))] = (_Float16)v;
}

// ------------------ 1) weight pre-swizzle + pad/zero ------------------
__global__ __launch_bounds__(256) void prep_kernel(const float* __restrict__ W1,
                                                   const float* __restrict__ W2,
                                                   _Float16* __restrict__ W1sw,
                                                   _Float16* __restrict__ W2sw,
                                                   _Float16* __restrict__ A2sw) {
    int idx = blockIdx.x * 256 + threadIdx.x;
    if (idx < 2048 * 208) {                       // W1: K=2048, N pad 200->208
        int k = idx / 208, n = idx % 208;
        float v = (n < 200) ? W1[k * 200 + n] : 0.0f;
        storeB(W1sw, n >> 4, n & 15, k, 64, v);
    }
    if (idx < 224 * 208) {                        // W2: K pad 200->224, N pad 200->208
        int k = idx / 208, n = idx % 208;
        float v = (k < 200 && n < 200) ? W2[k * 200 + n] : 0.0f;
        storeB(W2sw, n >> 4, n & 15, k, 7, v);
    }
    if (idx < 16 * 7 * 512) A2sw[idx] = (_Float16)0.0f;  // zero K-pad region for FC2 A
}

// ------------------ 2) ROI adaptive max-pool -> A1 fragments ------------------
__global__ __launch_bounds__(256) void roi_pool_kernel(const float* __restrict__ fm,
                                                       const int* __restrict__ coords,
                                                       _Float16* __restrict__ A1sw) {
    const int n = blockIdx.x;        // proposal
    const int t = threadIdx.x;       // channel pair (t, t+256)
    int y0 = coords[n * 4 + 0], x0 = coords[n * 4 + 1];
    int y1 = coords[n * 4 + 2], x1 = coords[n * 4 + 3];
    int sy = imin_(y0 >> 3, 74);     // H-2
    int sx = imin_(x0 >> 3, 126);    // W-2
    int ey = (y1 + 7) >> 3, ex = (x1 + 7) >> 3;
    int h = imax_(ey - sy, 2), w = imax_(ex - sx, 2);
    const int mt = n >> 4, m = n & 15;

    for (int i = 0; i < 2; ++i) {
        int rs = (i * h) >> 1, re = ((i + 1) * h + 1) >> 1;
        for (int j = 0; j < 2; ++j) {
            int cs = (j * w) >> 1, ce = ((j + 1) * w + 1) >> 1;
            float m0 = -1e30f, m1 = -1e30f;
            for (int r = rs; r < re; ++r) {
                const float* rowp = fm + (size_t)(sy + r) * 128 * 512;
                for (int cc = cs; cc < ce; ++cc) {
                    const float* p = rowp + (size_t)(sx + cc) * 512;
                    m0 = fmaxf(m0, p[t]);
                    m1 = fmaxf(m1, p[t + 256]);
                }
            }
            int kbase = (i * 2 + j) * 512;
            storeA(A1sw, mt, m, kbase + t, 64, m0);
            storeA(A1sw, mt, m, kbase + t + 256, 64, m1);
        }
    }
}

// ------------------ 3) FC1: [256,2048]x[2048,208] + bias + ReLU -> A2 fragments ---
__global__ __launch_bounds__(32) void fc1_kernel(const _Float16* __restrict__ A1sw,
                                                 const _Float16* __restrict__ W1sw,
                                                 const float* __restrict__ b1,
                                                 _Float16* __restrict__ A2sw) {
    const int mt = blockIdx.x, nt = blockIdx.y, lane = threadIdx.x;
    const v16h* Af = (const v16h*)A1sw;
    const v16h* Bf = (const v16h*)W1sw;
    v8f acc = {};
    for (int s = 0; s < 64; ++s) {
        v16h a = Af[(mt * 64 + s) * 32 + lane];
        v16h b = Bf[(nt * 64 + s) * 32 + lane];
        acc = __builtin_amdgcn_wmma_f32_16x16x32_f16(false, a, false, b,
                                                     (short)0, acc, false, false);
    }
    const int nl = lane & 15, hf = lane >> 4;
    const int n = nt * 16 + nl;              // K-index for FC2
#pragma unroll
    for (int r = 0; r < 8; ++r) {
        int ml = hf * 8 + r;                 // local row in 16x16 tile
        float v = 0.0f;
        if (n < 200) v = fmaxf(acc[r] + b1[n], 0.0f);
        storeA(A2sw, mt, ml, n, 7, v);
    }
}

// ------------------ 4) FC2: [256,224]x[224,208] + bias + ReLU -> F f32 -----------
__global__ __launch_bounds__(32) void fc2_kernel(const _Float16* __restrict__ A2sw,
                                                 const _Float16* __restrict__ W2sw,
                                                 const float* __restrict__ b2,
                                                 float* __restrict__ F) {
    const int mt = blockIdx.x, nt = blockIdx.y, lane = threadIdx.x;
    const v16h* Af = (const v16h*)A2sw;
    const v16h* Bf = (const v16h*)W2sw;
    v8f acc = {};
    for (int s = 0; s < 7; ++s) {
        v16h a = Af[(mt * 7 + s) * 32 + lane];
        v16h b = Bf[(nt * 7 + s) * 32 + lane];
        acc = __builtin_amdgcn_wmma_f32_16x16x32_f16(false, a, false, b,
                                                     (short)0, acc, false, false);
    }
    const int nl = lane & 15, hf = lane >> 4;
    const int n = nt * 16 + nl;
#pragma unroll
    for (int r = 0; r < 8; ++r) {
        int row = mt * 16 + hf * 8 + r;
        if (n < 200) F[row * 208 + n] = fmaxf(acc[r] + b2[n], 0.0f);
    }
}

// ------------------ 5) heads: FC3/FC4, argmax, softmax, box decode ---------------
__global__ __launch_bounds__(32) void head_kernel(const float* __restrict__ F,
                                                  const int* __restrict__ coords,
                                                  const float* __restrict__ W3,
                                                  const float* __restrict__ b3,
                                                  const float* __restrict__ W4,
                                                  const float* __restrict__ b4,
                                                  float* __restrict__ out) {
    const int n = blockIdx.x * 32 + threadIdx.x;
    if (n >= 256) return;
    float reg[16], cls[5];
#pragma unroll
    for (int j = 0; j < 16; ++j) reg[j] = b3[j];
#pragma unroll
    for (int j = 0; j < 5; ++j) cls[j] = b4[j];
    const float* f = F + n * 208;
    for (int k = 0; k < 200; ++k) {
        float fv = f[k];
        const float* w3r = W3 + k * 16;
        const float* w4r = W4 + k * 5;
#pragma unroll
        for (int j = 0; j < 16; ++j) reg[j] = fmaf(fv, w3r[j], reg[j]);
#pragma unroll
        for (int j = 0; j < 5; ++j) cls[j] = fmaf(fv, w4r[j], cls[j]);
    }
    // argmax (first-max, like jnp.argmax)
    int ci = 0;
    float best = cls[0];
#pragma unroll
    for (int j = 1; j < 5; ++j)
        if (cls[j] > best) { best = cls[j]; ci = j; }
    // softmax + score = max of non-bg probs
    float sum = 0.0f, e[5];
#pragma unroll
    for (int j = 0; j < 5; ++j) { e[j] = __expf(cls[j] - best); sum = sum + e[j]; }
    float score = 0.0f;
#pragma unroll
    for (int j = 1; j < 5; ++j) score = fmaxf(score, e[j] / sum);
    // select reg row for predicted class (bg uses clipped index)
    int rid = imax_(ci - 1, 0);
    float rr[4];
#pragma unroll
    for (int q = 0; q < 4; ++q) {
        float v = reg[q];
        if (rid == 1) v = reg[4 + q];
        if (rid == 2) v = reg[8 + q];
        if (rid == 3) v = reg[12 + q];
        rr[q] = v;
    }
    float y0 = (float)coords[n * 4 + 0], x0 = (float)coords[n * 4 + 1];
    float y1 = (float)coords[n * 4 + 2], x1 = (float)coords[n * 4 + 3];
    float ph = y1 - y0, pw = x1 - x0;
    float py = y0 + 0.5f * ph, px = x0 + 0.5f * pw;
    float p_y = fmaf(ph, rr[0], py);
    float p_x = fmaf(pw, rr[1], px);
    float p_h = ph * fminf(fmaxf(expf(rr[2]), 0.001f), 20.0f);
    float p_w = pw * fminf(fmaxf(expf(rr[3]), 0.001f), 20.0f);
    out[n * 4 + 0] = p_y;
    out[n * 4 + 1] = p_x;
    out[n * 4 + 2] = p_h;
    out[n * 4 + 3] = p_w;
    out[1024 + n] = score;
    out[1280 + n] = (ci != 0) ? 1.0f : 0.0f;
}

extern "C" void kernel_launch(void* const* d_in, const int* in_sizes, int n_in,
                              void* d_out, int out_size, void* d_ws, size_t ws_size,
                              hipStream_t stream) {
    const float* fm     = (const float*)d_in[0];   // [1,76,128,512]
    const int*   coords = (const int*)  d_in[1];   // [256,4]
    const float* W1     = (const float*)d_in[2];   // [2048,200]
    const float* b1     = (const float*)d_in[3];   // [200]
    const float* W2     = (const float*)d_in[4];   // [200,200]
    const float* b2     = (const float*)d_in[5];   // [200]
    const float* W3     = (const float*)d_in[6];   // [200,16]
    const float* b3     = (const float*)d_in[7];   // [16]
    const float* W4     = (const float*)d_in[8];   // [200,5]
    const float* b4     = (const float*)d_in[9];   // [5]
    float* out = (float*)d_out;

    char* ws = (char*)d_ws;
    _Float16* A1sw = (_Float16*)(ws + A1_OFF);
    _Float16* W1sw = (_Float16*)(ws + W1_OFF);
    _Float16* A2sw = (_Float16*)(ws + A2_OFF);
    _Float16* W2sw = (_Float16*)(ws + W2_OFF);
    float*    F    = (float*)   (ws + F_OFF);

    (void)in_sizes; (void)n_in; (void)out_size; (void)ws_size;

    // 1) swizzle weights into WMMA fragment layouts (and zero FC2 A pad region)
    prep_kernel<<<(2048 * 208 + 255) / 256, 256, 0, stream>>>(W1, W2, W1sw, W2sw, A2sw);
    // 2) ROI adaptive max-pool directly into FC1 A fragments (f16)
    roi_pool_kernel<<<256, 256, 0, stream>>>(fm, coords, A1sw);
    // 3) FC1 via WMMA f16->f32, fused bias+ReLU, emit FC2 A fragments
    fc1_kernel<<<dim3(16, 13), 32, 0, stream>>>(A1sw, W1sw, b1, A2sw);
    // 4) FC2 via WMMA, fused bias+ReLU -> f32 features
    fc2_kernel<<<dim3(16, 13), 32, 0, stream>>>(A2sw, W2sw, b2, F);
    // 5) classification/regression heads + box decode
    head_kernel<<<8, 32, 0, stream>>>(F, coords, W3, b3, W4, b4, out);
}
